// SharedAttentionProcessor_23373212025066
// MI455X (gfx1250) — matrix operand order repaired and checked by reading, not compiled, not run on hardware
//
#include <hip/hip_runtime.h>
#include <hip/hip_bf16.h>
#include <math.h>
#include <stdint.h>

// ---------------------------------------------------------------------------
// SharedAttentionProcessor for MI455X (gfx1250), bf16 WMMA pipeline.
// B=4, S=1024, C=1280, H=20, hd=64.
// Round 3: GEMM uses 32x64 tiles per wave with ping-pong (no-copy) double
// buffering; attention keeps async global->LDS staged K/V tiles.
// ---------------------------------------------------------------------------

typedef __bf16 bf16_t;
typedef __attribute__((ext_vector_type(16))) __bf16 v16bf;
typedef __attribute__((ext_vector_type(8)))  __bf16 v8bf;
typedef __attribute__((ext_vector_type(8)))  float  v8f;

union BF16x16 { v16bf v; v8bf h[2]; };

#define WB   4
#define WS   1024
#define WC   1280
#define WH   20
#define WHD  64
#define SHIFT_F 0.6931471805599453f
#define EPS_F   1e-5f

static __device__ __forceinline__ v8f wmma_bf16(v16bf a, v16bf b, v8f c) {
  return __builtin_amdgcn_wmma_f32_16x16x32_bf16(false, a, false, b,
                                                 (short)0, c, false, false);
}

// Async global->LDS copy, 16 bytes per lane (ASYNCcnt-tracked, no VGPR data).
// LDS aperture keeps the LDS offset in addr[31:0] (ISA 10.2), so the low
// 32 bits of the generic shared-pointer are the DS destination address.
static __device__ __forceinline__ void async_copy_b128(void* lds, const void* g) {
  uint32_t l = (uint32_t)(uintptr_t)lds;
  uint64_t a = (uint64_t)(uintptr_t)g;
  asm volatile("global_load_async_to_lds_b128 %0, %1, off"
               :: "v"(l), "v"(a) : "memory");
}
static __device__ __forceinline__ void wait_async0() {
  asm volatile("s_wait_asynccnt 0x0" ::: "memory");
}

// Reductions across one 16-lane half (xor masks < 16 stay within the half).
static __device__ __forceinline__ float half_max(float v) {
#pragma unroll
  for (int m = 1; m < 16; m <<= 1) v = fmaxf(v, __shfl_xor(v, m, 32));
  return v;
}
static __device__ __forceinline__ float half_sum(float v) {
#pragma unroll
  for (int m = 1; m < 16; m <<= 1) v += __shfl_xor(v, m, 32);
  return v;
}

// ---------------------------------------------------------------------------
// f32 -> bf16 conversion
// ---------------------------------------------------------------------------
__global__ void cvt_f32_bf16(const float* __restrict__ x,
                             bf16_t* __restrict__ y, int n) {
  int i = blockIdx.x * blockDim.x + threadIdx.x;
  if (i < n) y[i] = (bf16_t)x[i];
}

// ---------------------------------------------------------------------------
// GEMM: Out[M,N] = A[M,K] * W[N,K]^T (+bias). One wave -> 32x64 tile
// (two 16-row A tiles share every B fetch). K-loop unrolled by 2 with two
// named register sets: loads of set B issue between WMMAs of set A and vice
// versa -> latency hidden without register-rotation copies.
// ---------------------------------------------------------------------------
__global__ __launch_bounds__(256) void gemm_xwt(
    const bf16_t* __restrict__ A, const bf16_t* __restrict__ W,
    float* __restrict__ Out, const float* __restrict__ bias,
    int M, int N, int K) {
  const int lane = threadIdx.x & 31;
  const int half = lane >> 4;
  const int lq   = lane & 15;
  const int wave = (int)((blockIdx.x * blockDim.x + threadIdx.x) >> 5);
  const int nt64 = N >> 6;
  const int mt = wave / nt64;
  const int nt = wave % nt64;
  if (mt * 32 >= M) return;              // wave-uniform exit (EXEC stays full)
  const int m0 = mt * 32, n0 = nt * 64;

  const bf16_t* arow0 = A + (size_t)(m0 + lq) * K;
  const bf16_t* arow1 = A + (size_t)(m0 + 16 + lq) * K;

  auto loadA = [&](const bf16_t* arow, int kk, BF16x16& a) {
    a.h[0] = *(const v8bf*)(arow + kk + half * 16);
    a.h[1] = *(const v8bf*)(arow + kk + half * 16 + 8);
  };
  auto loadB = [&](int kk, v16bf* bm) {
#pragma unroll
    for (int t = 0; t < 4; ++t)
      bm[t] = *(const v16bf*)(W + (size_t)(n0 + t * 16 + lq) * K + kk + half * 16);
  };

  v8f acc0[4] = {};   // rows m0 .. m0+15
  v8f acc1[4] = {};   // rows m0+16 .. m0+31

  BF16x16 aA0, aA1, aB0, aB1;
  v16bf bA[4], bB[4];
  loadA(arow0, 0, aA0);
  loadA(arow1, 0, aA1);
  loadB(0, bA);

  for (int kk = 0; kk < K; kk += 64) {
    __builtin_prefetch(arow0 + kk + 512, 0, 1);    // global_prefetch_b8
    // set B loads (kk+32) then consume set A (kk)
    loadA(arow0, kk + 32, aB0);
    loadA(arow1, kk + 32, aB1);
    loadB(kk + 32, bB);
#pragma unroll
    for (int t = 0; t < 4; ++t) {
      acc0[t] = wmma_bf16(aA0.v, bA[t], acc0[t]);
      acc1[t] = wmma_bf16(aA1.v, bA[t], acc1[t]);
    }
    // set A loads (kk+64) then consume set B (kk+32)
    const int k2 = (kk + 64 < K) ? kk + 64 : kk;   // last iter: harmless reload
    loadA(arow0, k2, aA0);
    loadA(arow1, k2, aA1);
    loadB(k2, bA);
#pragma unroll
    for (int t = 0; t < 4; ++t) {
      acc0[t] = wmma_bf16(aB0.v, bB[t], acc0[t]);
      acc1[t] = wmma_bf16(aB1.v, bB[t], acc1[t]);
    }
  }

  // C/D layout: VGPR r -> row M = r + 8*half, col N = L%16.
#pragma unroll
  for (int t = 0; t < 4; ++t) {
    int n = n0 + t * 16 + lq;
    float badd = bias ? bias[n] : 0.0f;
#pragma unroll
    for (int r = 0; r < 8; ++r) {
      Out[(size_t)(m0 + r + half * 8) * N + n]      = acc0[t][r] + badd;
      Out[(size_t)(m0 + 16 + r + half * 8) * N + n] = acc1[t][r] + badd;
    }
  }
}

// ---------------------------------------------------------------------------
// AdaIN statistics over the sequence axis: mean/std per (b, c), ddof=1.
// ---------------------------------------------------------------------------
__global__ __launch_bounds__(256) void adain_stats(
    const float* __restrict__ X, float* __restrict__ mean,
    float* __restrict__ stdv) {
  __shared__ float s_sum[4][64];
  __shared__ float s_sq[4][64];
  const int chunks = WC / 64;
  int b  = blockIdx.x / chunks;
  int cb = (blockIdx.x % chunks) * 64;
  int tc = threadIdx.x & 63;
  int j  = threadIdx.x >> 6;
  const float* base = X + (size_t)b * WS * WC + cb + tc;
  float s = 0.f, sq = 0.f;
  for (int row = j; row < WS; row += 4) {
    float v = base[(size_t)row * WC];
    s += v; sq += v * v;
  }
  s_sum[j][tc] = s; s_sq[j][tc] = sq;
  __syncthreads();
  if (threadIdx.x < 64) {
    float ts = 0.f, tq = 0.f;
#pragma unroll
    for (int r = 0; r < 4; ++r) { ts += s_sum[r][threadIdx.x]; tq += s_sq[r][threadIdx.x]; }
    float mu  = ts / (float)WS;
    float var = (tq - ts * ts / (float)WS) / (float)(WS - 1);
    mean[b * WC + cb + threadIdx.x] = mu;
    stdv[b * WC + cb + threadIdx.x] = sqrtf(var + EPS_F);
  }
}

// AdaIN apply: y = (x - mean_b)/std_b * std_style + mean_style, -> bf16
__global__ void adain_apply(const float* __restrict__ X,
                            const float* __restrict__ mean,
                            const float* __restrict__ stdv,
                            bf16_t* __restrict__ Y) {
  size_t i = (size_t)blockIdx.x * blockDim.x + threadIdx.x;
  const size_t total = (size_t)WB * WS * WC;
  if (i >= total) return;
  int c = (int)(i % WC);
  int b = (int)(i / ((size_t)WS * WC));
  int sb = (b < WB / 2) ? 0 : WB / 2;
  float x = X[i];
  float y = (x - mean[b * WC + c]) / stdv[b * WC + c] * stdv[sb * WC + c] +
            mean[sb * WC + c];
  Y[i] = (bf16_t)y;
}

// V transpose+convert: [B,S,C] f32 -> [B,C,S] bf16.
__global__ void v_transpose(const float* __restrict__ X,
                            bf16_t* __restrict__ Y) {
  size_t i = (size_t)blockIdx.x * blockDim.x + threadIdx.x;
  const size_t total = (size_t)WB * WS * WC;
  if (i >= total) return;
  int c = (int)(i % WC);
  size_t bs = i / WC;
  int s = (int)(bs % WS);
  int b = (int)(bs / WS);
  Y[((size_t)b * WC + c) * WS + s] = (bf16_t)X[i];
}

// ---------------------------------------------------------------------------
// Flash-attention with shared-key concat + log(2) bias on the style block.
// All 8 waves in a block share one (b,h): K/V tiles (32 keys) are staged
// once per block into LDS with async b128 copies, double-buffered, so tile
// kt+1 streams in while tile kt feeds the WMMAs from LDS.
// ---------------------------------------------------------------------------
__global__ __launch_bounds__(256) void attn_kernel(
    const bf16_t* __restrict__ Q, const bf16_t* __restrict__ Kk,
    const bf16_t* __restrict__ Vt, bf16_t* __restrict__ O) {
  __shared__ bf16_t kst[2][32][64];   // [buf][key][d]     8 KB
  __shared__ bf16_t vst[2][64][32];   // [buf][d][key]     8 KB
  __shared__ bf16_t plds[8][16 * 32]; // per-wave P tile    8 KB

  const int lane = threadIdx.x & 31;
  const int half = lane >> 4;
  const int lq   = lane & 15;
  const int w    = threadIdx.x >> 5;
  int bid = blockIdx.x;
  int b   = bid / (WH * 8);
  int rem = bid % (WH * 8);
  int h   = rem / 8;
  int q0  = (rem % 8) * 128 + w * 16;
  int sbv = (b < WB / 2) ? 0 : WB / 2;
  const float scale = 0.125f;  // 1/sqrt(64)

  // Stage tile kt2 (32 keys of K and V) into buffer sbuf: 2 async b128/thread.
  auto stage = [&](int kt2, int sbuf) {
    const int kb2 = (kt2 < 32) ? b : sbv;
    const int kr0 = (kt2 & 31) * 32;
    {
      int row = threadIdx.x >> 3, ch = threadIdx.x & 7;   // 32 rows x 64 d
      const bf16_t* g = Kk + ((size_t)kb2 * WS + kr0 + row) * WC + h * WHD + ch * 8;
      async_copy_b128(&kst[sbuf][row][ch * 8], g);
    }
    {
      int dr = threadIdx.x >> 2, ch = threadIdx.x & 3;    // 64 d x 32 keys
      const bf16_t* g = Vt + ((size_t)(kb2 * WC + h * WHD + dr)) * WS + kr0 + ch * 8;
      async_copy_b128(&vst[sbuf][dr][ch * 8], g);
    }
  };

  // Preload Q A-tiles (two K=32 chunks covering hd=64).
  const bf16_t* qrow = Q + ((size_t)b * WS + q0 + lq) * WC + h * WHD;
  BF16x16 aq[2];
#pragma unroll
  for (int c = 0; c < 2; ++c) {
    aq[c].h[0] = *(const v8bf*)(qrow + c * 32 + half * 16);
    aq[c].h[1] = *(const v8bf*)(qrow + c * 32 + half * 16 + 8);
  }

  v8f o[4] = {};
  float mrow[8], lrow[8];
#pragma unroll
  for (int r = 0; r < 8; ++r) { mrow[r] = -INFINITY; lrow[r] = 0.f; }

  // Prologue: stage tile 0.
  stage(0, 0);
  wait_async0();
  __syncthreads();

  for (int kt = 0; kt < 64; ++kt) {
    const int pb = kt & 1;
    const float bias = (kt < 32) ? 0.f : SHIFT_F;

    // Kick off the next tile into the other buffer (safe: everyone passed the
    // barrier, so no wave still reads it).
    if (kt + 1 < 64) stage(kt + 1, pb ^ 1);

    // ---- S = Q K^T * scale + bias (two 16-key subtiles), B from LDS ----
    v8f s0 = {}, s1 = {};
#pragma unroll
    for (int c = 0; c < 2; ++c) {
      v16bf b0 = *(const v16bf*)&kst[pb][lq][c * 32 + half * 16];
      v16bf b1 = *(const v16bf*)&kst[pb][16 + lq][c * 32 + half * 16];
      s0 = wmma_bf16(aq[c].v, b0, s0);
      s1 = wmma_bf16(aq[c].v, b1, s1);
    }

    // ---- online softmax (row = r + 8*half, columns across the half) ----
    float p0[8], p1[8], corr[8];
#pragma unroll
    for (int r = 0; r < 8; ++r) {
      float v0 = s0[r] * scale + bias;
      float v1 = s1[r] * scale + bias;
      float rm = half_max(fmaxf(v0, v1));
      float mn = fmaxf(mrow[r], rm);
      float cr = __expf(mrow[r] - mn);
      float e0 = __expf(v0 - mn);
      float e1 = __expf(v1 - mn);
      float rs = half_sum(e0 + e1);
      lrow[r] = lrow[r] * cr + rs;
      mrow[r] = mn;
      corr[r] = cr;
      p0[r] = e0; p1[r] = e1;
    }
#pragma unroll
    for (int t = 0; t < 4; ++t)
#pragma unroll
      for (int r = 0; r < 8; ++r) o[t][r] *= corr[r];

    // ---- transpose P (D-layout) -> A-layout via per-wave LDS ----
#pragma unroll
    for (int r = 0; r < 8; ++r) {
      int qr = r + half * 8;
      plds[w][qr * 32 + lq]      = (bf16_t)p0[r];
      plds[w][qr * 32 + 16 + lq] = (bf16_t)p1[r];
    }
    asm volatile("s_wait_dscnt 0x0" ::: "memory");  // wave32 in-order DS
    BF16x16 ap;
    ap.h[0] = *(const v8bf*)&plds[w][lq * 32 + half * 16];
    ap.h[1] = *(const v8bf*)&plds[w][lq * 32 + half * 16 + 8];

    // ---- O += P * V (B from staged Vt tile in LDS) ----
#pragma unroll
    for (int t = 0; t < 4; ++t) {
      v16bf vb = *(const v16bf*)&vst[pb][t * 16 + lq][half * 16];
      o[t] = wmma_bf16(ap.v, vb, o[t]);
    }

    // Retire this iteration: our async copies done, all waves done reading.
    wait_async0();
    __syncthreads();
  }

  // ---- normalize and store bf16 [B,S,C] ----
  float linv[8];
#pragma unroll
  for (int r = 0; r < 8; ++r) linv[r] = 1.0f / lrow[r];
#pragma unroll
  for (int t = 0; t < 4; ++t)
#pragma unroll
    for (int r = 0; r < 8; ++r)
      O[((size_t)b * WS + q0 + r + half * 8) * WC + h * WHD + t * 16 + lq] =
          (bf16_t)(o[t][r] * linv[r]);
}

// ---------------------------------------------------------------------------
// Host orchestration
// ---------------------------------------------------------------------------
extern "C" void kernel_launch(void* const* d_in, const int* in_sizes, int n_in,
                              void* d_out, int out_size, void* d_ws, size_t ws_size,
                              hipStream_t stream) {
  (void)in_sizes; (void)n_in; (void)out_size; (void)ws_size;
  const float* X  = (const float*)d_in[0];
  const float* Wq = (const float*)d_in[1];
  const float* Wk = (const float*)d_in[2];
  const float* Wv = (const float*)d_in[3];
  const float* Wo = (const float*)d_in[4];
  const float* bo = (const float*)d_in[5];
  float* out = (float*)d_out;

  const size_t M  = (size_t)WB * WS;      // 4096 rows
  const size_t nX = M * WC;               // 5.24M elems
  const size_t nW = (size_t)WC * WC;

  char* p = (char*)d_ws;
  auto carve = [&](size_t bytes) -> char* {
    char* r = p; p += (bytes + 255) & ~(size_t)255; return r;
  };
  bf16_t* Xh  = (bf16_t*)carve(nX * 2);
  bf16_t* Wqh = (bf16_t*)carve(nW * 2);
  bf16_t* Wkh = (bf16_t*)carve(nW * 2);
  bf16_t* Wvh = (bf16_t*)carve(nW * 2);
  bf16_t* Woh = (bf16_t*)carve(nW * 2);
  float*  Qf  = (float*)carve(nX * 4);
  float*  Kf  = (float*)carve(nX * 4);
  float*  Vf  = (float*)carve(nX * 4);
  bf16_t* Qh  = (bf16_t*)carve(nX * 2);
  bf16_t* Kh  = (bf16_t*)carve(nX * 2);
  bf16_t* Vt  = (bf16_t*)carve(nX * 2);
  bf16_t* Ah  = (bf16_t*)carve(nX * 2);
  float* meanQ = (float*)carve((size_t)WB * WC * 4);
  float* stdQ  = (float*)carve((size_t)WB * WC * 4);
  float* meanK = (float*)carve((size_t)WB * WC * 4);
  float* stdK  = (float*)carve((size_t)WB * WC * 4);

  const int T = 256;
  const int gX = (int)((nX + T - 1) / T);
  const int gW = (int)((nW + T - 1) / T);

  // 0) f32 -> bf16
  cvt_f32_bf16<<<gX, T, 0, stream>>>(X,  Xh,  (int)nX);
  cvt_f32_bf16<<<gW, T, 0, stream>>>(Wq, Wqh, (int)nW);
  cvt_f32_bf16<<<gW, T, 0, stream>>>(Wk, Wkh, (int)nW);
  cvt_f32_bf16<<<gW, T, 0, stream>>>(Wv, Wvh, (int)nW);
  cvt_f32_bf16<<<gW, T, 0, stream>>>(Wo, Woh, (int)nW);

  // 1) Q/K/V projections: (4096/32)*(1280/64) = 2560 waves -> 320 blocks
  const int gemmBlocks = (int)((M / 32) * (WC / 64) * 32 / T);
  gemm_xwt<<<gemmBlocks, T, 0, stream>>>(Xh, Wqh, Qf, nullptr, (int)M, WC, WC);
  gemm_xwt<<<gemmBlocks, T, 0, stream>>>(Xh, Wkh, Kf, nullptr, (int)M, WC, WC);
  gemm_xwt<<<gemmBlocks, T, 0, stream>>>(Xh, Wvh, Vf, nullptr, (int)M, WC, WC);

  // 2) AdaIN on Q and K; V transposed+converted
  adain_stats<<<WB * (WC / 64), T, 0, stream>>>(Qf, meanQ, stdQ);
  adain_stats<<<WB * (WC / 64), T, 0, stream>>>(Kf, meanK, stdK);
  adain_apply<<<gX, T, 0, stream>>>(Qf, meanQ, stdQ, Qh);
  adain_apply<<<gX, T, 0, stream>>>(Kf, meanK, stdK, Kh);
  v_transpose<<<gX, T, 0, stream>>>(Vf, Vt);

  // 3) shared-key flash attention: B*H*8 = 640 blocks
  attn_kernel<<<WB * WH * 8, T, 0, stream>>>(Qh, Kh, Vt, Ah);

  // 4) output projection with bias -> f32 d_out
  gemm_xwt<<<gemmBlocks, T, 0, stream>>>(Ah, Woh, out, bo, (int)M, WC, WC);
}
